// RSTAttBlock2_73065983640209
// MI455X (gfx1250) — matrix operand order repaired and checked by reading, compiled, not run on hardware
//
#include <hip/hip_runtime.h>
#include <cstdint>
#include <cstddef>

// ---- problem constants ----
#define BB 32
#define TT 12
#define NN 1024
#define DD 64
#define FF 16
#define KHN 8
#define DHN 8
#define BT (BB * TT)                 // 384
#define NTOK ((size_t)BT * NN)       // 393216 tokens, each 64 wide

typedef __attribute__((ext_vector_type(16))) __bf16 v16bf;
typedef __attribute__((ext_vector_type(8)))  __bf16 v8bf;
typedef __attribute__((ext_vector_type(8)))  float  v8f;

// exact pointee type from the hipcc diagnostic for the async-LDS builtin
typedef int async_v4i __attribute__((vector_size(4 * sizeof(int))));

// packed-weight (W^T) column offsets inside WT[64][1024]
#define WOFF_GCN 0
#define WOFF_Q   256
#define WOFF_K   384
#define WOFF_V   512
#define WOFF_X1  640
#define WOFF_X2  704
#define WOFF_S   768
#define WOFF_T   832
#define WOFF_H1  896
#define WOFF_H2  960

#if defined(__has_builtin)
#if __has_builtin(__builtin_amdgcn_global_load_async_to_lds_b128) && \
    __has_builtin(__builtin_amdgcn_s_wait_asynccnt)
#define AGG_ASYNC 1
#endif
#endif

// ---- bf16 helpers (RNE) ----
__device__ __forceinline__ unsigned short f2bfu(float f) {
  unsigned u = __builtin_bit_cast(unsigned, f);
  unsigned r = u + 0x7fffu + ((u >> 16) & 1u);
  return (unsigned short)(r >> 16);
}
__device__ __forceinline__ float bfu2f(unsigned short h) {
  return __builtin_bit_cast(float, (unsigned)h << 16);
}
__device__ __forceinline__ v8f v8f_zero() {
  v8f v;
#pragma unroll
  for (int i = 0; i < 8; ++i) v[i] = 0.0f;
  return v;
}

// ---- wave-level GEMM: 16 rows x 64 cols, bf16 operands, f32 accumulate ----
// ptrA(m, k) -> const ushort* to 8 contiguous bf16 of A row m starting at k
// ptrB(n, k) -> const ushort* to 8 contiguous bf16 of B^T row n starting at k
template <typename PA, typename PB>
__device__ __forceinline__ void wmma_gemm_16x64(PA ptrA, PB ptrB, int K, v8f acc[4]) {
  const int lane = threadIdx.x & 31;
  const int lm = lane & 15;
  const int half = lane >> 4;
  for (int k0 = 0; k0 < K; k0 += 32) {
    v8bf a0 = *(const v8bf*)ptrA(lm, k0 + half * 8);
    v8bf a1 = *(const v8bf*)ptrA(lm, k0 + 16 + half * 8);
    v16bf a = __builtin_shufflevector(a0, a1, 0, 1, 2, 3, 4, 5, 6, 7,
                                      8, 9, 10, 11, 12, 13, 14, 15);
#pragma unroll
    for (int nt = 0; nt < 4; ++nt) {
      const int n = nt * 16 + lm;
      v8bf b0 = *(const v8bf*)ptrB(n, k0 + half * 8);
      v8bf b1 = *(const v8bf*)ptrB(n, k0 + 16 + half * 8);
      v16bf b = __builtin_shufflevector(b0, b1, 0, 1, 2, 3, 4, 5, 6, 7,
                                        8, 9, 10, 11, 12, 13, 14, 15);
      acc[nt] = __builtin_amdgcn_wmma_f32_16x16x32_bf16(
          false, a, false, b, (short)0, acc[nt], false, false);
    }
  }
}

// store callback st(m, n, value) with m in [0,16), n in [0,64)
template <typename FS>
__device__ __forceinline__ void store_16x64(const v8f acc[4], FS st) {
  const int lane = threadIdx.x & 31;
  const int ln = lane & 15;
  const int half = lane >> 4;
#pragma unroll
  for (int nt = 0; nt < 4; ++nt)
#pragma unroll
    for (int j = 0; j < 8; ++j)
      st(j + 8 * half, nt * 16 + ln, acc[nt][j]);
}

// =====================================================================
// P1: plain fp32 -> bf16 conversion (STE, adj)
// =====================================================================
__global__ void cvt1d_kernel(const float* __restrict__ src,
                             unsigned short* __restrict__ dst, size_t n) {
  size_t i = ((size_t)blockIdx.x * blockDim.x + threadIdx.x) * 4;
  if (i < n) {
    float4 v = *(const float4*)(src + i);
    dst[i + 0] = f2bfu(v.x);
    dst[i + 1] = f2bfu(v.y);
    dst[i + 2] = f2bfu(v.z);
    dst[i + 3] = f2bfu(v.w);
  }
}

// =====================================================================
// P2: X -> Xbf (row major bf16) and XT (per-(b,t) transposed bf16 [64][1024])
// =====================================================================
__global__ void xpose_kernel(const float* __restrict__ X,
                             unsigned short* __restrict__ Xbf,
                             unsigned short* __restrict__ XT) {
  __shared__ float tile[64][65];
  const int n0 = blockIdx.x * 64;
  const int bt = blockIdx.y;
  const int c = threadIdx.x & 63;
  const int r4 = threadIdx.x >> 6;   // 0..3
#pragma unroll
  for (int i = 0; i < 16; ++i) {
    int row = i * 4 + r4;
    float v = X[((size_t)bt * NN + n0 + row) * DD + c];
    Xbf[((size_t)bt * NN + n0 + row) * DD + c] = f2bfu(v);
    tile[row][c] = v;
  }
  __syncthreads();
#pragma unroll
  for (int i = 0; i < 16; ++i) {
    int d = i * 4 + r4;
    XT[((size_t)bt * DD + d) * NN + n0 + c] = f2bfu(tile[c][d]);
  }
}

// =====================================================================
// P3: pack all weights transposed into WT[64][1024] bf16
// =====================================================================
__global__ void wcvt_kernel(const float* __restrict__ Wgcn, const float* __restrict__ Wq,
                            const float* __restrict__ Wk, const float* __restrict__ Wv,
                            const float* __restrict__ Wx1, const float* __restrict__ Wx2,
                            const float* __restrict__ Ws, const float* __restrict__ Wt,
                            const float* __restrict__ Wh1, const float* __restrict__ Wh2,
                            unsigned short* __restrict__ WT) {
  int idx = blockIdx.x * blockDim.x + threadIdx.x;  // 64*1024
  int n = idx >> 10, kk = idx & 1023;
  const float* W; int k;
  if      (kk < 256) { W = Wgcn; k = kk; }
  else if (kk < 384) { W = Wq;  k = kk - 256; }
  else if (kk < 512) { W = Wk;  k = kk - 384; }
  else if (kk < 640) { W = Wv;  k = kk - 512; }
  else if (kk < 704) { W = Wx1; k = kk - 640; }
  else if (kk < 768) { W = Wx2; k = kk - 704; }
  else if (kk < 832) { W = Ws;  k = kk - 768; }
  else if (kk < 896) { W = Wt;  k = kk - 832; }
  else if (kk < 960) { W = Wh1; k = kk - 896; }
  else               { W = Wh2; k = kk - 960; }
  WT[(size_t)n * 1024 + kk] = f2bfu(W[(size_t)k * DD + n]);
}

// =====================================================================
// K1: FEbf[which][N][64] = bf16(relu(feat @ W)); FEQ (which 0,2) pre-scaled
// by 1/sqrt(64)=0.125 (exact power of two) so smat needs no post-scale.
// =====================================================================
__global__ void fe_kernel(const float* __restrict__ feat0, const float* __restrict__ feat1,
                          const float* __restrict__ Wq0, const float* __restrict__ Wk0,
                          const float* __restrict__ Wq1, const float* __restrict__ Wk1,
                          unsigned short* __restrict__ FEbf) {
  int idx = blockIdx.x * blockDim.x + threadIdx.x;   // 4*1024*64
  int which = idx >> 16;
  int rem = idx & 65535;
  int n = rem >> 6, c = rem & 63;
  const float* feat = (which < 2) ? feat0 : feat1;
  const float* W = (which == 0) ? Wq0 : (which == 1) ? Wk0 : (which == 2) ? Wq1 : Wk1;
  float s = 0.0f;
#pragma unroll
  for (int f = 0; f < FF; ++f) s += feat[n * FF + f] * W[f * DD + c];
  s = fmaxf(s, 0.0f);
  if ((which & 1) == 0) s *= 0.125f;   // fold attention scale into FEQ
  FEbf[idx] = f2bfu(s);
}

// =====================================================================
// K2: S[g] = softmax(FEQ' @ FEK^T) row-wise, bf16 out. 16 rows / block.
// =====================================================================
__global__ void smat_kernel(const unsigned short* __restrict__ FEbf,
                            unsigned short* __restrict__ S) {
  __shared__ float sc[16 * NN];   // 64 KB score tile
  const int mt = blockIdx.x;
  const int g = blockIdx.y;
  const unsigned short* FEQ = FEbf + (size_t)(2 * g) * NN * DD;
  const unsigned short* FEK = FEbf + (size_t)(2 * g + 1) * NN * DD;
  const int wid = threadIdx.x >> 5;
  const int lane = threadIdx.x & 31;
  const int m0 = mt * 16;
  for (int ct = 0; ct < 4; ++ct) {
    const int n0 = wid * 256 + ct * 64;
    v8f acc[4] = {v8f_zero(), v8f_zero(), v8f_zero(), v8f_zero()};
    wmma_gemm_16x64(
        [&](int m, int k) { return FEQ + (size_t)(m0 + m) * DD + k; },
        [&](int n, int k) { return FEK + (size_t)(n0 + n) * DD + k; },
        DD, acc);
    store_16x64(acc, [&](int m, int n, float v) { sc[m * NN + n0 + n] = v; });
  }
  __syncthreads();
  for (int r = wid * 4; r < wid * 4 + 4; ++r) {
    float mx = -3.0e38f;
    for (int c = lane; c < NN; c += 32) mx = fmaxf(mx, sc[r * NN + c]);
#pragma unroll
    for (int off = 16; off; off >>= 1) mx = fmaxf(mx, __shfl_xor(mx, off));
    float sum = 0.0f;
    for (int c = lane; c < NN; c += 32) {
      float e = __expf(sc[r * NN + c] - mx);
      sc[r * NN + c] = e;
      sum += e;
    }
#pragma unroll
    for (int off = 16; off; off >>= 1) sum += __shfl_xor(sum, off);
    float inv = 1.0f / sum;
    for (int c = lane; c < NN; c += 32)
      S[(size_t)g * NN * NN + (size_t)(m0 + r) * NN + c] = f2bfu(sc[r * NN + c] * inv);
  }
}

// =====================================================================
// K3: aggregation Y[which][bt] = M @ X[bt]; M in {adjbf, S0, S1}; K = 1024
// grid (8, BT, 3), block 256 (8 waves, one 16-row tile each).
// All 8 waves share the B panel (X^T chunk): stage it in LDS with
// async DMA (GLOBAL_LOAD_ASYNC_TO_LDS_B128), double-buffered.
// =====================================================================
__global__ void agg_kernel(const unsigned short* __restrict__ XT,
                           const unsigned short* __restrict__ adjbf,
                           const unsigned short* __restrict__ S,
                           unsigned short* __restrict__ Yagg) {
  const int wid = threadIdx.x >> 5;
  const int lane = threadIdx.x & 31;
  const int lm = lane & 15;
  const int half = lane >> 4;
  const int m0 = (blockIdx.x * 8 + wid) * 16;
  const int bt = blockIdx.y;
  const int which = blockIdx.z;
  const unsigned short* A =
      (which == 0) ? adjbf : S + (size_t)(which - 1) * NN * NN;
  const unsigned short* XTbt = XT + (size_t)bt * DD * NN;
  v8f acc[4] = {v8f_zero(), v8f_zero(), v8f_zero(), v8f_zero()};

#if defined(AGG_ASYNC)
  // double-buffered B panel: [2][64 rows][32 k] bf16 = 8 KB
  __shared__ __align__(16) unsigned short Bp[2][64 * 32];
  const int srow = threadIdx.x >> 2;        // 0..63 : B^T row n
  const int sseg = (threadIdx.x & 3) * 8;   // k segment within chunk
  auto issue = [&](int chunk, int buf) {
    const unsigned short* src = XTbt + (size_t)srow * NN + chunk * 32 + sseg;
    unsigned short* dst = &Bp[buf][srow * 32 + sseg];
    __builtin_amdgcn_global_load_async_to_lds_b128(
        (__attribute__((address_space(1))) async_v4i*)src,
        (__attribute__((address_space(3))) async_v4i*)dst, 0, 0);
  };
  const int NC = NN / 32;
  issue(0, 0);
  for (int i = 0; i < NC; ++i) {
    const int buf = i & 1;
    if (i + 1 < NC) {
      issue(i + 1, buf ^ 1);
      __builtin_amdgcn_s_wait_asynccnt(1);   // chunk i arrived (in-order)
    } else {
      __builtin_amdgcn_s_wait_asynccnt(0);
    }
    __syncthreads();                          // panel visible to all waves
    const int k0 = i * 32;
    v8bf a0 = *(const v8bf*)(A + (size_t)(m0 + lm) * NN + k0 + half * 8);
    v8bf a1 = *(const v8bf*)(A + (size_t)(m0 + lm) * NN + k0 + 16 + half * 8);
    v16bf a = __builtin_shufflevector(a0, a1, 0, 1, 2, 3, 4, 5, 6, 7,
                                      8, 9, 10, 11, 12, 13, 14, 15);
#pragma unroll
    for (int nt = 0; nt < 4; ++nt) {
      const unsigned short* bp = &Bp[buf][(nt * 16 + lm) * 32 + half * 8];
      v8bf b0 = *(const v8bf*)bp;
      v8bf b1 = *(const v8bf*)(bp + 16);
      v16bf b = __builtin_shufflevector(b0, b1, 0, 1, 2, 3, 4, 5, 6, 7,
                                        8, 9, 10, 11, 12, 13, 14, 15);
      acc[nt] = __builtin_amdgcn_wmma_f32_16x16x32_bf16(
          false, a, false, b, (short)0, acc[nt], false, false);
    }
    __syncthreads();                          // done reading before overwrite
  }
#else
  wmma_gemm_16x64(
      [&](int m, int k) { return A + (size_t)(m0 + m) * NN + k; },
      [&](int n, int k) { return XTbt + (size_t)n * NN + k; },
      NN, acc);
#endif

  unsigned short* Y = Yagg + ((size_t)which * NTOK + (size_t)bt * NN) * DD;
  store_16x64(acc, [&](int m, int n, float v) { Y[(size_t)(m0 + m) * DD + n] = f2bfu(v); });
}

// =====================================================================
// K4: HS = relu(concat(Xbf, Yadj, Ys0, Ys1) @ Wgcn + bgcn), K = 256
// =====================================================================
__global__ void gcn_kernel(const unsigned short* __restrict__ Xbf,
                           const unsigned short* __restrict__ Yagg,
                           const unsigned short* __restrict__ WT,
                           const float* __restrict__ bgcn,
                           unsigned short* __restrict__ HS) {
  const int wid = threadIdx.x >> 5;
  const size_t r0 = ((size_t)blockIdx.x * 8 + wid) * 16;
  const unsigned short* Y0 = Yagg;
  const unsigned short* Y1 = Yagg + (size_t)NTOK * DD;
  const unsigned short* Y2 = Yagg + (size_t)2 * NTOK * DD;
  v8f acc[4] = {v8f_zero(), v8f_zero(), v8f_zero(), v8f_zero()};
  wmma_gemm_16x64(
      [&](int m, int k) {
        const unsigned short* base =
            (k < 64) ? Xbf : (k < 128) ? Y0 : (k < 192) ? Y1 : Y2;
        return base + (r0 + m) * DD + (k & 63);
      },
      [&](int n, int k) { return WT + (size_t)n * 1024 + WOFF_GCN + k; },
      4 * DD, acc);
  store_16x64(acc, [&](int m, int n, float v) {
    HS[(r0 + m) * DD + n] = f2bfu(fmaxf(v + bgcn[n], 0.0f));
  });
}

// =====================================================================
// K5: QKV[p] = relu(concat(Xbf, STEbf) @ W[p] + b[p]), K = 128
// =====================================================================
__global__ void qkv_kernel(const unsigned short* __restrict__ Xbf,
                           const unsigned short* __restrict__ STEbf,
                           const unsigned short* __restrict__ WT,
                           const float* __restrict__ bq, const float* __restrict__ bk,
                           const float* __restrict__ bv,
                           unsigned short* __restrict__ QKV) {
  const int p = blockIdx.z;
  const float* bias = (p == 0) ? bq : (p == 1) ? bk : bv;
  const int woff = WOFF_Q + p * 128;
  const int wid = threadIdx.x >> 5;
  const size_t r0 = ((size_t)blockIdx.x * 8 + wid) * 16;
  v8f acc[4] = {v8f_zero(), v8f_zero(), v8f_zero(), v8f_zero()};
  wmma_gemm_16x64(
      [&](int m, int k) {
        const unsigned short* base = (k < 64) ? Xbf : STEbf;
        return base + (r0 + m) * DD + (k & 63);
      },
      [&](int n, int k) { return WT + (size_t)n * 1024 + woff + k; },
      2 * DD, acc);
  unsigned short* out = QKV + (size_t)p * NTOK * DD;
  store_16x64(acc, [&](int m, int n, float v) {
    out[(r0 + m) * DD + n] = f2bfu(fmaxf(v + bias[n], 0.0f));
  });
}

// =====================================================================
// K6: temporal attention over T=12, one thread per (b,t,n,head); VALU only
// =====================================================================
__device__ __forceinline__ void ld8bf(const unsigned short* p, float* f) {
  uint4 u = *(const uint4*)p;
  unsigned w0 = u.x, w1 = u.y, w2 = u.z, w3 = u.w;
  f[0] = bfu2f((unsigned short)(w0 & 0xffff)); f[1] = bfu2f((unsigned short)(w0 >> 16));
  f[2] = bfu2f((unsigned short)(w1 & 0xffff)); f[3] = bfu2f((unsigned short)(w1 >> 16));
  f[4] = bfu2f((unsigned short)(w2 & 0xffff)); f[5] = bfu2f((unsigned short)(w2 >> 16));
  f[6] = bfu2f((unsigned short)(w3 & 0xffff)); f[7] = bfu2f((unsigned short)(w3 >> 16));
}

__global__ void attn_kernel(const unsigned short* __restrict__ QKV,
                            unsigned short* __restrict__ O) {
  const unsigned short* Q = QKV;
  const unsigned short* Kp = QKV + (size_t)NTOK * DD;
  const unsigned short* V = QKV + (size_t)2 * NTOK * DD;
  size_t idx = (size_t)blockIdx.x * blockDim.x + threadIdx.x;  // NTOK * KH
  const int h = (int)(idx & 7);
  const int n = (int)((idx >> 3) & (NN - 1));
  const int bt = (int)(idx >> 13);
  const int b = bt / TT;
  const size_t qbase = ((size_t)bt * NN + n) * DD + h * DHN;
  float q[DHN];
  ld8bf(Q + qbase, q);
  float sc[TT];
  float mx = -3.0e38f;
#pragma unroll
  for (int s = 0; s < TT; ++s) {
    size_t base = (((size_t)b * TT + s) * NN + n) * DD + h * DHN;
    float kr[DHN];
    ld8bf(Kp + base, kr);
    float dot = 0.0f;
#pragma unroll
    for (int d = 0; d < DHN; ++d) dot += q[d] * kr[d];
    sc[s] = dot * 0.3535533906f;   // 1/sqrt(8)
    mx = fmaxf(mx, sc[s]);
  }
  float sum = 0.0f;
#pragma unroll
  for (int s = 0; s < TT; ++s) { sc[s] = __expf(sc[s] - mx); sum += sc[s]; }
  const float inv = 1.0f / sum;
  float o[DHN];
#pragma unroll
  for (int d = 0; d < DHN; ++d) o[d] = 0.0f;
#pragma unroll
  for (int s = 0; s < TT; ++s) {
    size_t base = (((size_t)b * TT + s) * NN + n) * DD + h * DHN;
    float vr[DHN];
    ld8bf(V + base, vr);
    float p = sc[s] * inv;
#pragma unroll
    for (int d = 0; d < DHN; ++d) o[d] += p * vr[d];
  }
#pragma unroll
  for (int d = 0; d < DHN; ++d) O[qbase + d] = f2bfu(o[d]);
}

// =====================================================================
// K7: HT = relu(O @ Wx1 + bx1) @ Wx2 + bx2, fused via per-wave bf16 LDS tile
// =====================================================================
__global__ void ht_kernel(const unsigned short* __restrict__ O,
                          const unsigned short* __restrict__ WT,
                          const float* __restrict__ bx1, const float* __restrict__ bx2,
                          unsigned short* __restrict__ HT) {
  __shared__ __align__(16) unsigned short u[8 * 16 * DD];   // 16 KB
  const int wid = threadIdx.x >> 5;
  unsigned short* uw = u + wid * 16 * DD;
  const size_t r0 = ((size_t)blockIdx.x * 8 + wid) * 16;
  v8f acc[4] = {v8f_zero(), v8f_zero(), v8f_zero(), v8f_zero()};
  wmma_gemm_16x64(
      [&](int m, int k) { return O + (r0 + m) * DD + k; },
      [&](int n, int k) { return WT + (size_t)n * 1024 + WOFF_X1 + k; },
      DD, acc);
  store_16x64(acc, [&](int m, int n, float v) {
    uw[m * DD + n] = f2bfu(fmaxf(v + bx1[n], 0.0f));
  });
  __syncthreads();
  v8f acc2[4] = {v8f_zero(), v8f_zero(), v8f_zero(), v8f_zero()};
  wmma_gemm_16x64(
      [&](int m, int k) { return (const unsigned short*)(uw + m * DD + k); },
      [&](int n, int k) { return WT + (size_t)n * 1024 + WOFF_X2 + k; },
      DD, acc2);
  store_16x64(acc2, [&](int m, int n, float v) {
    HT[(r0 + m) * DD + n] = f2bfu(v + bx2[n]);
  });
}

// =====================================================================
// K8: gated fusion + output MLP + residual, fully fused per 16-row tile
// =====================================================================
__global__ void fuse_kernel(const float* __restrict__ X,
                            const unsigned short* __restrict__ HS,
                            const unsigned short* __restrict__ HT,
                            const unsigned short* __restrict__ WT,
                            const float* __restrict__ btb,
                            const float* __restrict__ bh1, const float* __restrict__ bh2,
                            float* __restrict__ out) {
  __shared__ __align__(16) unsigned short hm[8 * 16 * DD];  // 16 KB
  const int wid = threadIdx.x >> 5;
  const int lane = threadIdx.x & 31;
  const int ln = lane & 15;
  const int half = lane >> 4;
  unsigned short* hw = hm + wid * 16 * DD;
  const size_t r0 = ((size_t)blockIdx.x * 8 + wid) * 16;

  v8f a1[4] = {v8f_zero(), v8f_zero(), v8f_zero(), v8f_zero()};
  wmma_gemm_16x64(
      [&](int m, int k) { return HS + (r0 + m) * DD + k; },
      [&](int n, int k) { return WT + (size_t)n * 1024 + WOFF_S + k; },
      DD, a1);
  v8f a2[4] = {v8f_zero(), v8f_zero(), v8f_zero(), v8f_zero()};
  wmma_gemm_16x64(
      [&](int m, int k) { return HT + (r0 + m) * DD + k; },
      [&](int n, int k) { return WT + (size_t)n * 1024 + WOFF_T + k; },
      DD, a2);

  // zg = sigmoid(HS@Ws + HT@Wt + bt); Hm = zg*HS + (1-zg)*HT  -> LDS (bf16)
#pragma unroll
  for (int nt = 0; nt < 4; ++nt)
#pragma unroll
    for (int j = 0; j < 8; ++j) {
      int m = j + 8 * half, n = nt * 16 + ln;
      float g = a1[nt][j] + a2[nt][j] + btb[n];
      float z = 1.0f / (1.0f + __expf(-g));
      size_t i = (r0 + m) * DD + n;
      float hs = bfu2f(HS[i]);
      float ht = bfu2f(HT[i]);
      hw[m * DD + n] = f2bfu(z * hs + (1.0f - z) * ht);
    }
  __syncthreads();

  v8f a3[4] = {v8f_zero(), v8f_zero(), v8f_zero(), v8f_zero()};
  wmma_gemm_16x64(
      [&](int m, int k) { return (const unsigned short*)(hw + m * DD + k); },
      [&](int n, int k) { return WT + (size_t)n * 1024 + WOFF_H1 + k; },
      DD, a3);
  __syncthreads();
  store_16x64(a3, [&](int m, int n, float v) {
    hw[m * DD + n] = f2bfu(fmaxf(v + bh1[n], 0.0f));
  });
  __syncthreads();

  v8f a4[4] = {v8f_zero(), v8f_zero(), v8f_zero(), v8f_zero()};
  wmma_gemm_16x64(
      [&](int m, int k) { return (const unsigned short*)(hw + m * DD + k); },
      [&](int n, int k) { return WT + (size_t)n * 1024 + WOFF_H2 + k; },
      DD, a4);
  store_16x64(a4, [&](int m, int n, float v) {
    size_t i = (r0 + m) * DD + n;
    out[i] = X[i] + v + bh2[n];
  });
}

// =====================================================================
extern "C" void kernel_launch(void* const* d_in, const int* in_sizes, int n_in,
                              void* d_out, int out_size, void* d_ws, size_t ws_size,
                              hipStream_t stream) {
  (void)in_sizes; (void)n_in; (void)out_size; (void)ws_size;
  const float* X    = (const float*)d_in[0];
  const float* STE  = (const float*)d_in[1];
  const float* adj  = (const float*)d_in[2];
  const float* feat0= (const float*)d_in[3];
  const float* feat1= (const float*)d_in[4];
  const float* Wq0  = (const float*)d_in[5];
  const float* Wk0  = (const float*)d_in[6];
  const float* Wq1  = (const float*)d_in[7];
  const float* Wk1  = (const float*)d_in[8];
  const float* Wgcn = (const float*)d_in[9];
  const float* bgcn = (const float*)d_in[10];
  const float* Wq   = (const float*)d_in[11];
  const float* bq   = (const float*)d_in[12];
  const float* Wk   = (const float*)d_in[13];
  const float* bk   = (const float*)d_in[14];
  const float* Wv   = (const float*)d_in[15];
  const float* bv   = (const float*)d_in[16];
  const float* Wx1  = (const float*)d_in[17];
  const float* bx1  = (const float*)d_in[18];
  const float* Wx2  = (const float*)d_in[19];
  const float* bx2  = (const float*)d_in[20];
  const float* Ws   = (const float*)d_in[21];
  const float* Wt   = (const float*)d_in[22];
  const float* btb  = (const float*)d_in[23];
  const float* Wh1  = (const float*)d_in[24];
  const float* bh1  = (const float*)d_in[25];
  const float* Wh2  = (const float*)d_in[26];
  const float* bh2  = (const float*)d_in[27];
  float* out = (float*)d_out;

  // ---- workspace layout (bf16 intermediates), with buffer reuse ----
  char* ws = (char*)d_ws;
  size_t off = 0;
  unsigned short* Smat  = (unsigned short*)(ws + off); off += 2ull * NN * NN * 2;      // 4 MiB
  unsigned short* Yagg  = (unsigned short*)(ws + off); off += 3ull * NTOK * DD * 2;    // 144 MiB
  unsigned short* HSb   = (unsigned short*)(ws + off); off += (size_t)NTOK * DD * 2;   // 48 MiB
  unsigned short* Xbf   = (unsigned short*)(ws + off); off += (size_t)NTOK * DD * 2;   // 48 MiB
  unsigned short* STEbf = (unsigned short*)(ws + off); off += (size_t)NTOK * DD * 2;   // 48 MiB
  unsigned short* XT    = (unsigned short*)(ws + off); off += (size_t)NTOK * DD * 2;   // 48 MiB
  unsigned short* adjbf = (unsigned short*)(ws + off); off += (size_t)NN * NN * 2;     // 2 MiB
  unsigned short* FEbf  = (unsigned short*)(ws + off); off += 4ull * NN * DD * 2;      // 0.5 MiB
  unsigned short* WTp   = (unsigned short*)(ws + off); off += 64ull * 1024 * 2;        // 128 KiB
  unsigned short* QKV = Yagg;    // reuse after gcn consumed Yagg
  unsigned short* Ob  = XT;      // reuse after agg consumed XT
  unsigned short* HTb = STEbf;   // reuse after qkv consumed STEbf

  const unsigned tiles = (unsigned)(NTOK / 128);   // 3072 blocks of 8 waves

  wcvt_kernel <<<dim3(64 * 1024 / 256), dim3(256), 0, stream>>>(
      Wgcn, Wq, Wk, Wv, Wx1, Wx2, Ws, Wt, Wh1, Wh2, WTp);
  fe_kernel   <<<dim3(4 * NN * DD / 256), dim3(256), 0, stream>>>(
      feat0, feat1, Wq0, Wk0, Wq1, Wk1, FEbf);
  cvt1d_kernel<<<dim3((unsigned)(NN * NN / 4 / 256)), dim3(256), 0, stream>>>(
      adj, adjbf, (size_t)NN * NN);
  cvt1d_kernel<<<dim3((unsigned)(NTOK * DD / 4 / 256)), dim3(256), 0, stream>>>(
      STE, STEbf, NTOK * DD);
  xpose_kernel<<<dim3(NN / 64, BT), dim3(256), 0, stream>>>(X, Xbf, XT);
  smat_kernel <<<dim3(NN / 16, 2), dim3(128), 0, stream>>>(FEbf, Smat);
  agg_kernel  <<<dim3(8, BT, 3), dim3(256), 0, stream>>>(XT, adjbf, Smat, Yagg);
  gcn_kernel  <<<dim3(tiles), dim3(256), 0, stream>>>(Xbf, Yagg, WTp, bgcn, HSb);
  qkv_kernel  <<<dim3(tiles, 1, 3), dim3(256), 0, stream>>>(Xbf, STEbf, WTp, bq, bk, bv, QKV);
  attn_kernel <<<dim3((unsigned)(NTOK * KHN / 256)), dim3(256), 0, stream>>>(QKV, Ob);
  ht_kernel   <<<dim3(tiles), dim3(256), 0, stream>>>(Ob, WTp, bx1, bx2, HTb);
  fuse_kernel <<<dim3(tiles), dim3(256), 0, stream>>>(X, HSb, HTb, WTp, btb, bh1, bh2, out);
}
